// charEmbedder_53936199303290
// MI455X (gfx1250) — matrix operand order repaired and checked by reading, not compiled, not run on hardware
//
#include <hip/hip_runtime.h>
#include <cmath>

// GEMM: out[M=8192, N=512] = tanh( X[8192, K=8192] * W^T ), W is [512, 8192] row-major.
// bf16 WMMA (V_WMMA_F32_16X16X32_BF16), f32 accumulate.
//  - 32x512 output tile per WG: 8 waves x (2 M-tiles x 4 N-tiles) -> 8 WMMAs per k-step
//  - A/B fragment double buffering pinned with sched_barrier(0); B additionally
//    pipelined ACROSS chunk boundaries (no L2 stall at the chunk head)
//  - LDS double buffering for the X f32->bf16 staging (one barrier per K-chunk)

typedef __attribute__((ext_vector_type(8)))  float   v8f;
typedef __attribute__((ext_vector_type(8)))  __bf16  bf16x8;
typedef __attribute__((ext_vector_type(16))) __bf16  bf16x16;

#define M_TOTAL 8192   // B*S = 16*512
#define K_TOTAL 8192   // V
#define N_TOTAL 512    // D
#define M_TILE  32     // rows per workgroup (2 x 16 per wave)
#define K_CHUNK 128
#define KSTEPS  (K_CHUNK / 32)
#define LDS_STRIDE 136 // 128 + 8 bf16 pad -> spreads LDS banks (272B rows, 16B aligned)

// ---------------- W f32 -> bf16 conversion pre-pass ----------------
__global__ __launch_bounds__(256) void conv_w_bf16(const float* __restrict__ W,
                                                   __bf16* __restrict__ Wb) {
    const int i = (blockIdx.x * 256 + threadIdx.x) * 8;
    const float4 f0 = *(const float4*)(W + i);
    const float4 f1 = *(const float4*)(W + i + 4);
    bf16x8 o;
    o[0] = (__bf16)f0.x; o[1] = (__bf16)f0.y; o[2] = (__bf16)f0.z; o[3] = (__bf16)f0.w;
    o[4] = (__bf16)f1.x; o[5] = (__bf16)f1.y; o[6] = (__bf16)f1.z; o[7] = (__bf16)f1.w;
    *(bf16x8*)(Wb + i) = o;
}

// A fragment (16x32 bf16) from LDS, contiguous K order (matches B):
// lanes 0-15: M=lane, K=0..15; lanes 16-31: K=16..31.
static __device__ __forceinline__ bf16x16 load_a(const __bf16* sb, int mt, int colLane,
                                                 int khalf, int ks) {
    const __bf16* ap = sb + (mt * 16 + colLane) * LDS_STRIDE + ks * 32 + khalf * 16;
    const bf16x8 a0 = *(const bf16x8*)(ap);
    const bf16x8 a1 = *(const bf16x8*)(ap + 8);
    return __builtin_shufflevector(a0, a1, 0, 1, 2, 3, 4, 5, 6, 7,
                                   8, 9, 10, 11, 12, 13, 14, 15);
}

static __device__ __forceinline__ void cvt16(const float4 f0, const float4 f1,
                                             const float4 f2, const float4 f3,
                                             __bf16* dst) {
    bf16x8 lo, hi;
    lo[0] = (__bf16)f0.x; lo[1] = (__bf16)f0.y; lo[2] = (__bf16)f0.z; lo[3] = (__bf16)f0.w;
    lo[4] = (__bf16)f1.x; lo[5] = (__bf16)f1.y; lo[6] = (__bf16)f1.z; lo[7] = (__bf16)f1.w;
    hi[0] = (__bf16)f2.x; hi[1] = (__bf16)f2.y; hi[2] = (__bf16)f2.z; hi[3] = (__bf16)f2.w;
    hi[4] = (__bf16)f3.x; hi[5] = (__bf16)f3.y; hi[6] = (__bf16)f3.z; hi[7] = (__bf16)f3.w;
    *(bf16x8*)(dst)     = lo;
    *(bf16x8*)(dst + 8) = hi;
}

// ---------------- main WMMA kernel ----------------
__global__ __launch_bounds__(256) void char_embed_wmma(const float* __restrict__ x,
                                                       const __bf16* __restrict__ Wb,
                                                       float* __restrict__ out) {
    __shared__ __align__(16) __bf16 smA[2][M_TILE * LDS_STRIDE];

    const int tid  = threadIdx.x;
    const int lane = tid & 31;
    const int wave = tid >> 5;              // 0..7 -> N columns [wave*64, wave*64+64)
    const int m0   = blockIdx.x * M_TILE;   // 32-row M tile

    // cooperative X staging: thread -> (row 0..31, 16 consecutive k)
    const int ldRow = tid >> 3;
    const int ldK   = (tid & 7) * 16;
    const float* xrow = x + (size_t)(m0 + ldRow) * K_TOTAL + ldK;
    const int    sOff = ldRow * LDS_STRIDE + ldK;

    // WMMA lane decomposition
    const int colLane = lane & 15;          // A: row M; B/C: column n
    const int khalf   = lane >> 4;          // K sub-range select

    // Per-lane B column base pointers (B[k,n] = Wb[n*K + k], contiguous 32B per frag).
    const __bf16* wcol[4];
    #pragma unroll
    for (int nt = 0; nt < 4; ++nt)
        wcol[nt] = Wb + (size_t)(wave * 64 + nt * 16 + colLane) * K_TOTAL + khalf * 16;

    v8f acc[2][4];
    #pragma unroll
    for (int mt = 0; mt < 2; ++mt)
        #pragma unroll
        for (int nt = 0; nt < 4; ++nt) acc[mt][nt] = v8f{};

    // B fragments live ACROSS chunks: loaded one k-step ahead, incl. over the
    // chunk boundary (B has no dependence on the LDS staging barrier).
    bf16x16 bC[4];
    #pragma unroll
    for (int nt = 0; nt < 4; ++nt) bC[nt] = *(const bf16x16*)(wcol[nt]);

    // Pipelined compute over one staged K-chunk (KSTEPS k-steps of 32).
    // prefetch_next: in the last k-step, prefetch B for the NEXT chunk's ks=0.
    auto compute_chunk = [&](int kb, const __bf16* sb, bool prefetch_next) {
        bf16x16 aC[2];
        #pragma unroll
        for (int mt = 0; mt < 2; ++mt) aC[mt] = load_a(sb, mt, colLane, khalf, 0);
        __builtin_amdgcn_sched_barrier(0);

        #pragma unroll
        for (int ks = 0; ks < KSTEPS; ++ks) {
            bf16x16 aN[2], bN[4];
            const bool midStep = (ks < KSTEPS - 1);
            if (midStep) {                      // prefetch next step's fragments
                #pragma unroll
                for (int mt = 0; mt < 2; ++mt)
                    aN[mt] = load_a(sb, mt, colLane, khalf, ks + 1);
                #pragma unroll
                for (int nt = 0; nt < 4; ++nt)
                    bN[nt] = *(const bf16x16*)(wcol[nt] + kb + (ks + 1) * 32);
            } else if (prefetch_next) {         // prefetch next CHUNK's first B
                #pragma unroll
                for (int nt = 0; nt < 4; ++nt)
                    bN[nt] = *(const bf16x16*)(wcol[nt] + kb + K_CHUNK);
            }
            __builtin_amdgcn_sched_barrier(0);  // keep prefetch loads above the WMMAs
            #pragma unroll
            for (int mt = 0; mt < 2; ++mt)
                #pragma unroll
                for (int nt = 0; nt < 4; ++nt)
                    acc[mt][nt] = __builtin_amdgcn_wmma_f32_16x16x32_bf16(
                        false, aC[mt], false, bC[nt], (short)0, acc[mt][nt],
                        false, false);
            __builtin_amdgcn_sched_barrier(0);  // keep WMMAs above the rotate copies
            if (midStep) {
                #pragma unroll
                for (int mt = 0; mt < 2; ++mt) aC[mt] = aN[mt];
            }
            if (midStep || prefetch_next) {
                #pragma unroll
                for (int nt = 0; nt < 4; ++nt) bC[nt] = bN[nt];
            }
        }
    };

    // Prologue: stage chunk 0 into buffer 0.
    {
        const float4* xp = (const float4*)(xrow);
        cvt16(xp[0], xp[1], xp[2], xp[3], &smA[0][sOff]);
    }
    __syncthreads();

    int buf = 0;
    for (int kb = 0; kb + K_CHUNK < K_TOTAL; kb += K_CHUNK) {
        // Issue next chunk's X loads early (overlap with compute below).
        const float4* xp = (const float4*)(xrow + kb + K_CHUNK);
        const float4 g0 = xp[0], g1 = xp[1], g2 = xp[2], g3 = xp[3];
        if (kb + 2 * K_CHUNK < K_TOTAL)
            __builtin_prefetch(xrow + kb + 2 * K_CHUNK, 0, 0);  // global_prefetch_b8
        __builtin_amdgcn_sched_barrier(0);  // keep X loads above the compute section

        compute_chunk(kb, smA[buf], /*prefetch_next=*/true);

        cvt16(g0, g1, g2, g3, &smA[buf ^ 1][sOff]);  // stage into the other buffer
        __syncthreads();
        buf ^= 1;
    }
    compute_chunk(K_TOTAL - K_CHUNK, smA[buf], /*prefetch_next=*/false);  // peeled tail

    // Epilogue: tanh + store. C layout: vgpr v, lanes 0-15 -> row v, lanes 16-31 -> row v+8.
    const int rowOff = khalf * 8;
    #pragma unroll
    for (int mt = 0; mt < 2; ++mt) {
        #pragma unroll
        for (int nt = 0; nt < 4; ++nt) {
            const int col = wave * 64 + nt * 16 + colLane;
            #pragma unroll
            for (int v = 0; v < 8; ++v) {
                const int row = m0 + mt * 16 + rowOff + v;
                out[(size_t)row * N_TOTAL + col] = tanhf(acc[mt][nt][v]);
            }
        }
    }
}

extern "C" void kernel_launch(void* const* d_in, const int* in_sizes, int n_in,
                              void* d_out, int out_size, void* d_ws, size_t ws_size,
                              hipStream_t stream) {
    (void)in_sizes; (void)n_in; (void)out_size; (void)ws_size;
    const float* x = (const float*)d_in[0];   // [16, 512, 8192] f32
    const float* W = (const float*)d_in[1];   // [512, 8192] f32
    float* out     = (float*)d_out;           // [16, 512, 512] f32
    __bf16* Wb     = (__bf16*)d_ws;           // 512*8192*2 = 8 MB scratch

    // W -> bf16 (4,194,304 elements, 8 per thread)
    conv_w_bf16<<<(N_TOTAL * K_TOTAL) / (256 * 8), 256, 0, stream>>>(W, Wb);

    // 256 M-tiles of 32 rows; 8 waves per WG cover all 512 N columns
    char_embed_wmma<<<M_TOTAL / M_TILE, 256, 0, stream>>>(x, Wb, out);
}
